// LlamaSparseMoeBlock_42056319763010
// MI455X (gfx1250) — compile-verified, matrix operand order checked by reading, not run on hardware
//
#include <hip/hip_runtime.h>
#include <hip/hip_bf16.h>

#define H_DIM 1024
#define F_DIM 2048
#define N_EXP 8
#define USE_ASYNC 1

typedef __attribute__((ext_vector_type(16))) __bf16 v16bf;
typedef __attribute__((ext_vector_type(8)))  float  v8f;

union Frag { v16bf v; unsigned int u[8]; };

__device__ __forceinline__ unsigned short f2bf(float f) {
    unsigned int u = __float_as_uint(f);
    u += 0x7FFFu + ((u >> 16) & 1u);   // round-to-nearest-even
    return (unsigned short)(u >> 16);
}

__device__ __forceinline__ unsigned lds_off(const void* p) {
    // low 32 bits of a flat LDS address are the wave-relative LDS byte address
    return (unsigned)(uintptr_t)p;
}

#if USE_ASYNC
__device__ __forceinline__ void async_b128(unsigned lds_addr, const void* gptr) {
    asm volatile("global_load_async_to_lds_b128 %0, %1, off"
                 :: "v"(lds_addr), "v"((unsigned long long)(uintptr_t)gptr)
                 : "memory");
}
__device__ __forceinline__ void async_wait0() {
    asm volatile("s_wait_asynccnt 0x0" ::: "memory");
}
#else
__device__ __forceinline__ void async_b128(unsigned lds_addr, const void* gptr) {
    // synchronous fallback: flat copy of 16 bytes into LDS
    uint4 d = *(const uint4*)gptr;
    *(uint4*)((uintptr_t)lds_addr + (((uintptr_t)-1) ^ 0xffffffffull ?
        ((uintptr_t)(__builtin_amdgcn_groupstaticsize()) * 0) : 0)) = d; // not used
}
__device__ __forceinline__ void async_wait0() {}
#endif

// ---------------------------------------------------------------- zero output + counters
__global__ void zero_out_kernel(float* __restrict__ out, int n4, int* __restrict__ cnt) {
    int i = blockIdx.x * blockDim.x + threadIdx.x;
    if (i < n4) {
        float4 z = {0.f, 0.f, 0.f, 0.f};
        ((float4*)out)[i] = z;
    }
    if (blockIdx.x == 0 && threadIdx.x < N_EXP) cnt[threadIdx.x] = 0;
}

// ---------------------------------------------------------------- x fp32 -> bf16 (layout preserved)
__global__ void cvt_bf16_kernel(const float* __restrict__ x, unsigned short* __restrict__ xb, int n) {
    int i = (blockIdx.x * blockDim.x + threadIdx.x) * 4;
    if (i < n) {
        float4 v = *((const float4*)(x + i));
        ushort4 o;
        o.x = f2bf(v.x); o.y = f2bf(v.y); o.z = f2bf(v.z); o.w = f2bf(v.w);
        *((ushort4*)(xb + i)) = o;
    }
}

// ---------------------------------------------------------------- weight fp32 [R][C] -> bf16 [C][R] (per expert in z)
__global__ void __launch_bounds__(256) transpose_cvt_kernel(
    const float* __restrict__ src, unsigned short* __restrict__ dst, int R, int C)
{
    __shared__ float tile[32][33];
    const float* s = src + (size_t)blockIdx.z * R * C;
    unsigned short* d = dst + (size_t)blockIdx.z * R * C;
    const int c0 = blockIdx.x * 32, r0 = blockIdx.y * 32;
    const int tx = threadIdx.x & 31, ty = threadIdx.x >> 5;  // 32 x 8
#pragma unroll
    for (int i = ty; i < 32; i += 8)
        tile[i][tx] = s[(size_t)(r0 + i) * C + c0 + tx];
    __syncthreads();
#pragma unroll
    for (int i = ty; i < 32; i += 8)
        d[(size_t)(c0 + i) * R + r0 + tx] = f2bf(tile[tx][i]);
}

// ---------------------------------------------------------------- router: logits, softmax, top-2, gather lists
__global__ void __launch_bounds__(256) router_kernel(
    const float* __restrict__ x, const float* __restrict__ gw,
    float* __restrict__ logits_out, float* __restrict__ w_slot,
    int* __restrict__ idx_list, int* __restrict__ cnt, int T)
{
    __shared__ float sgw[N_EXP * H_DIM];
    for (int i = threadIdx.x; i < N_EXP * H_DIM; i += 256) sgw[i] = gw[i];
    __syncthreads();

    const int wave = threadIdx.x >> 5, lane = threadIdx.x & 31;
    const int tok = blockIdx.x * 8 + wave;
    if (tok >= T) return;

    float acc[N_EXP];
#pragma unroll
    for (int e = 0; e < N_EXP; e++) acc[e] = 0.f;

    const float* xr = x + (size_t)tok * H_DIM;
    for (int h = lane; h < H_DIM; h += 32) {
        float xv = xr[h];
#pragma unroll
        for (int e = 0; e < N_EXP; e++) acc[e] += xv * sgw[e * H_DIM + h];
    }
#pragma unroll
    for (int e = 0; e < N_EXP; e++) {
        float v = acc[e];
        for (int off = 16; off > 0; off >>= 1) v += __shfl_xor(v, off, 32);
        acc[e] = v;
    }
    if (lane == 0) {
#pragma unroll
        for (int e = 0; e < N_EXP; e++) logits_out[(size_t)tok * N_EXP + e] = acc[e];
        int e1 = 0; float m1 = acc[0];
#pragma unroll
        for (int e = 1; e < N_EXP; e++) if (acc[e] > m1) { m1 = acc[e]; e1 = e; }
        int e2 = -1; float m2 = -3.4e38f;
#pragma unroll
        for (int e = 0; e < N_EXP; e++) if (e != e1 && acc[e] > m2) { m2 = acc[e]; e2 = e; }
        float p2 = __expf(m2 - m1);
        float inv = __builtin_amdgcn_rcpf(1.f + p2);   // top-k renorm; softmax normalizer cancels
        int s1 = atomicAdd(&cnt[e1], 1);
        idx_list[e1 * T + s1] = tok; w_slot[e1 * T + s1] = inv;
        int s2 = atomicAdd(&cnt[e2], 1);
        idx_list[e2 * T + s2] = tok; w_slot[e2 * T + s2] = p2 * inv;
    }
}

// ---------------------------------------------------------------- GEMM1: g = silu(x@Wg) * (x@Wu)  (gathered rows)
__global__ void __launch_bounds__(256) moe_gemm1(
    const unsigned short* __restrict__ xb,    // [T,H] bf16
    const unsigned short* __restrict__ WgT,   // [F,H] bf16 (pre-transposed)
    const unsigned short* __restrict__ WuT,   // [F,H] bf16
    unsigned short* __restrict__ gws,         // [T,F] bf16 intermediate
    const int* __restrict__ idx_list,
    const int* __restrict__ cnt_p)
{
    __shared__ unsigned short lA [2][128 * 40];
    __shared__ unsigned short lBg[2][ 64 * 40];
    __shared__ unsigned short lBu[2][ 64 * 40];

    const int cnt = *cnt_p;
    const int m0 = blockIdx.y * 128;
    if (m0 >= cnt) return;
    const int n0 = blockIdx.x * 64;
    const int tid = threadIdx.x;

    // A loader: 2 threads per row, 2x16B each
    const int arow = tid >> 1, ahalf = tid & 1;
    const int slot = m0 + arow;
    const int tokenA = (slot < cnt) ? idx_list[slot] : 0;
    const unsigned short* aSrc = xb + (size_t)tokenA * H_DIM + ahalf * 16;
    // B loader: 4 threads per row (64 rows), 16B each
    const int brow = tid >> 2, bchunk = (tid & 3) * 8;
    const unsigned short* bgSrc = WgT + (size_t)(n0 + brow) * H_DIM + bchunk;
    const unsigned short* buSrc = WuT + (size_t)(n0 + brow) * H_DIM + bchunk;

    unsigned aDst[2], bgDst[2], buDst[2];
#pragma unroll
    for (int b = 0; b < 2; b++) {
        aDst[b]  = lds_off(&lA [b][arow * 40 + ahalf * 16]);
        bgDst[b] = lds_off(&lBg[b][brow * 40 + bchunk]);
        buDst[b] = lds_off(&lBu[b][brow * 40 + bchunk]);
    }

    const int wave = tid >> 5, lane = tid & 31;
    const int wm = wave & 3, wn = wave >> 2;
    const int lm = lane & 15, kb = (lane >> 4) << 3;
    int kidx[8];
#pragma unroll
    for (int j = 0; j < 8; j++) kidx[j] = (j < 4) ? (kb + 2 * j) : (16 + kb + 2 * (j - 4));

    v8f zero = {0.f, 0.f, 0.f, 0.f, 0.f, 0.f, 0.f, 0.f};
    v8f accg[2][2], accu[2][2];
#pragma unroll
    for (int i = 0; i < 2; i++)
#pragma unroll
        for (int j = 0; j < 2; j++) { accg[i][j] = zero; accu[i][j] = zero; }

    auto stage = [&](int b, int k0) {
        async_b128(aDst[b],      aSrc + k0);
        async_b128(aDst[b] + 16, aSrc + k0 + 8);
        async_b128(bgDst[b],     bgSrc + k0);
        async_b128(buDst[b],     buSrc + k0);
    };

    int buf = 0;
    stage(0, 0);
    for (int k0 = 0; k0 < H_DIM; k0 += 32) {
        async_wait0();        // own async batch for `buf` landed
        __syncthreads();      // everyone's landed
        if (k0 + 32 < H_DIM) stage(buf ^ 1, k0 + 32);

        Frag a[2], bg[2], bu[2];
#pragma unroll
        for (int sm = 0; sm < 2; sm++) {
            int m = wm * 32 + sm * 16 + lm;
#pragma unroll
            for (int j = 0; j < 8; j++) a[sm].u[j] = *(const unsigned int*)&lA[buf][m * 40 + kidx[j]];
        }
#pragma unroll
        for (int sn = 0; sn < 2; sn++) {
            int n = wn * 32 + sn * 16 + lm;
#pragma unroll
            for (int j = 0; j < 8; j++) {
                bg[sn].u[j] = *(const unsigned int*)&lBg[buf][n * 40 + kidx[j]];
                bu[sn].u[j] = *(const unsigned int*)&lBu[buf][n * 40 + kidx[j]];
            }
        }
#pragma unroll
        for (int sm = 0; sm < 2; sm++)
#pragma unroll
            for (int sn = 0; sn < 2; sn++) {
                accg[sm][sn] = __builtin_amdgcn_wmma_f32_16x16x32_bf16(
                    false, a[sm].v, false, bg[sn].v, (short)0, accg[sm][sn], false, false);
                accu[sm][sn] = __builtin_amdgcn_wmma_f32_16x16x32_bf16(
                    false, a[sm].v, false, bu[sn].v, (short)0, accu[sm][sn], false, false);
            }
        buf ^= 1;
    }

    const int mhalf = (lane >> 4) << 3;
#pragma unroll
    for (int sm = 0; sm < 2; sm++)
#pragma unroll
        for (int sn = 0; sn < 2; sn++) {
#pragma unroll
            for (int r = 0; r < 8; r++) {
                int row = m0 + wm * 32 + sm * 16 + mhalf + r;
                int col = n0 + wn * 32 + sn * 16 + lm;
                float g = accg[sm][sn][r];
                float u = accu[sm][sn][r];
                float s = g * __builtin_amdgcn_rcpf(1.f + __expf(-g));   // silu
                gws[(size_t)row * F_DIM + col] = f2bf(s * u);
            }
        }
}

// ---------------------------------------------------------------- GEMM2: out[token] += w * (g @ Wd)
__global__ void __launch_bounds__(256) moe_gemm2(
    const unsigned short* __restrict__ gws,   // [T,F] bf16
    const unsigned short* __restrict__ WdT,   // [H,F] bf16 (pre-transposed)
    float* __restrict__ out,                  // [T,H] fp32 (atomic accumulate)
    const int* __restrict__ idx_list,
    const float* __restrict__ w_slot,
    const int* __restrict__ cnt_p)
{
    __shared__ unsigned short lA[2][128 * 40];
    __shared__ unsigned short lB[2][ 64 * 40];

    const int cnt = *cnt_p;
    const int m0 = blockIdx.y * 128;
    if (m0 >= cnt) return;
    const int n0 = blockIdx.x * 64;
    const int tid = threadIdx.x;

    const int arow = tid >> 1, ahalf = tid & 1;
    const unsigned short* aSrc = gws + (size_t)(m0 + arow) * F_DIM + ahalf * 16;
    const int brow = tid >> 2, bchunk = (tid & 3) * 8;
    const unsigned short* bSrc = WdT + (size_t)(n0 + brow) * F_DIM + bchunk;

    unsigned aDst[2], bDst[2];
#pragma unroll
    for (int b = 0; b < 2; b++) {
        aDst[b] = lds_off(&lA[b][arow * 40 + ahalf * 16]);
        bDst[b] = lds_off(&lB[b][brow * 40 + bchunk]);
    }

    const int wave = tid >> 5, lane = tid & 31;
    const int wm = wave & 3, wn = wave >> 2;
    const int lm = lane & 15, kb = (lane >> 4) << 3;
    int kidx[8];
#pragma unroll
    for (int j = 0; j < 8; j++) kidx[j] = (j < 4) ? (kb + 2 * j) : (16 + kb + 2 * (j - 4));

    v8f zero = {0.f, 0.f, 0.f, 0.f, 0.f, 0.f, 0.f, 0.f};
    v8f acc[2][2];
#pragma unroll
    for (int i = 0; i < 2; i++)
#pragma unroll
        for (int j = 0; j < 2; j++) acc[i][j] = zero;

    auto stage = [&](int b, int k0) {
        async_b128(aDst[b],      aSrc + k0);
        async_b128(aDst[b] + 16, aSrc + k0 + 8);
        async_b128(bDst[b],      bSrc + k0);
    };

    int buf = 0;
    stage(0, 0);
    for (int k0 = 0; k0 < F_DIM; k0 += 32) {
        async_wait0();
        __syncthreads();
        if (k0 + 32 < F_DIM) stage(buf ^ 1, k0 + 32);

        Frag a[2], b[2];
#pragma unroll
        for (int sm = 0; sm < 2; sm++) {
            int m = wm * 32 + sm * 16 + lm;
#pragma unroll
            for (int j = 0; j < 8; j++) a[sm].u[j] = *(const unsigned int*)&lA[buf][m * 40 + kidx[j]];
        }
#pragma unroll
        for (int sn = 0; sn < 2; sn++) {
            int n = wn * 32 + sn * 16 + lm;
#pragma unroll
            for (int j = 0; j < 8; j++) b[sn].u[j] = *(const unsigned int*)&lB[buf][n * 40 + kidx[j]];
        }
#pragma unroll
        for (int sm = 0; sm < 2; sm++)
#pragma unroll
            for (int sn = 0; sn < 2; sn++)
                acc[sm][sn] = __builtin_amdgcn_wmma_f32_16x16x32_bf16(
                    false, a[sm].v, false, b[sn].v, (short)0, acc[sm][sn], false, false);
        buf ^= 1;
    }

    const int mhalf = (lane >> 4) << 3;
#pragma unroll
    for (int sm = 0; sm < 2; sm++) {
#pragma unroll
        for (int r = 0; r < 8; r++) {
            int row = m0 + wm * 32 + sm * 16 + mhalf + r;
            if (row < cnt) {
                int tok = idx_list[row];
                float w = w_slot[row];
#pragma unroll
                for (int sn = 0; sn < 2; sn++) {
                    int col = n0 + wn * 32 + sn * 16 + lm;
                    atomicAdd(&out[(size_t)tok * H_DIM + col], w * acc[sm][sn][r]);
                }
            }
        }
    }
}

// ---------------------------------------------------------------- host launcher
extern "C" void kernel_launch(void* const* d_in, const int* in_sizes, int n_in,
                              void* d_out, int out_size, void* d_ws, size_t ws_size,
                              hipStream_t stream) {
    const float* x  = (const float*)d_in[0];   // [B,S,H]
    const float* gw = (const float*)d_in[1];   // [E,H]
    const float* Wg = (const float*)d_in[2];   // [E,H,F]
    const float* Wu = (const float*)d_in[3];   // [E,H,F]
    const float* Wd = (const float*)d_in[4];   // [E,F,H]

    const int T = in_sizes[0] / H_DIM;         // 8192

    float* out    = (float*)d_out;             // [T,H]
    float* logits = out + (size_t)T * H_DIM;   // [T,E]

    char* ws = (char*)d_ws;
    size_t o = 0;
    unsigned short* xb  = (unsigned short*)(ws + o); o += (size_t)T * H_DIM * 2;       // 16 MB
    unsigned short* gws = (unsigned short*)(ws + o); o += (size_t)T * F_DIM * 2;       // 32 MB
    unsigned short* WgT = (unsigned short*)(ws + o); o += (size_t)N_EXP * H_DIM * F_DIM * 2; // 32 MB
    unsigned short* WuT = (unsigned short*)(ws + o); o += (size_t)N_EXP * H_DIM * F_DIM * 2; // 32 MB
    unsigned short* WdT = (unsigned short*)(ws + o); o += (size_t)N_EXP * H_DIM * F_DIM * 2; // 32 MB
    float* w_slot       = (float*)(ws + o);          o += (size_t)N_EXP * T * 4;
    int*   idx_list     = (int*)(ws + o);            o += (size_t)N_EXP * T * 4;
    int*   cnt          = (int*)(ws + o);

    const int nOut = T * H_DIM;
    zero_out_kernel<<<(nOut / 4 + 255) / 256, 256, 0, stream>>>(out, nOut / 4, cnt);
    cvt_bf16_kernel<<<(nOut / 4 + 255) / 256, 256, 0, stream>>>(x, xb, nOut);
    // weights: [H,F] -> [F,H] bf16 for Wg/Wu; [F,H] -> [H,F] bf16 for Wd
    transpose_cvt_kernel<<<dim3(F_DIM / 32, H_DIM / 32, N_EXP), 256, 0, stream>>>(Wg, WgT, H_DIM, F_DIM);
    transpose_cvt_kernel<<<dim3(F_DIM / 32, H_DIM / 32, N_EXP), 256, 0, stream>>>(Wu, WuT, H_DIM, F_DIM);
    transpose_cvt_kernel<<<dim3(H_DIM / 32, F_DIM / 32, N_EXP), 256, 0, stream>>>(Wd, WdT, F_DIM, H_DIM);
    router_kernel<<<(T + 7) / 8, 256, 0, stream>>>(x, gw, logits, w_slot, idx_list, cnt, T);

    for (int e = 0; e < N_EXP; e++) {
        moe_gemm1<<<dim3(F_DIM / 64, T / 128), 256, 0, stream>>>(
            xb, WgT + (size_t)e * H_DIM * F_DIM, WuT + (size_t)e * H_DIM * F_DIM,
            gws, idx_list + (size_t)e * T, cnt + e);
        moe_gemm2<<<dim3(H_DIM / 64, T / 128), 256, 0, stream>>>(
            gws, WdT + (size_t)e * H_DIM * F_DIM, out,
            idx_list + (size_t)e * T, w_slot + (size_t)e * T, cnt + e);
    }
}
